// GAT_9096740733076
// MI455X (gfx1250) — compile-verified, matrix operand order checked by reading, not run on hardware
//
#include <hip/hip_runtime.h>

// ---------------------------------------------------------------------------
// Sizes fixed by the reference
// ---------------------------------------------------------------------------
#define DD 256      // entity/relation/head dim
#define HH 4        // heads
#define NN 200      // neighbors
#define BB 512      // batch
#define NT 208      // neighbor rows materialized in LDS (13 tiles)
#define CH 64       // chunk rows per flash step (4 tiles, uniform)
#define NCHUNK 4    // 4*64 = 256 logical rows; rows >= 200 are masked out
#define HUGE_I 1e31f

typedef unsigned short u16;
typedef __attribute__((ext_vector_type(16))) __bf16 bf16x16;
typedef __attribute__((ext_vector_type(8)))  float  f32x8;

__device__ __forceinline__ u16 f2bf(float x) {            // RNE f32 -> bf16
  unsigned u = __float_as_uint(x);
  u += 0x7FFFu + ((u >> 16) & 1u);
  return (u16)(u >> 16);
}
__device__ __forceinline__ float bf2f(u16 h) {
  return __uint_as_float(((unsigned)h) << 16);
}

// ---------------------------------------------------------------------------
// WMMA bf16 fragment loader.
// A-fragment (16x32, row-major [row][k], stride DD halves):
//   lane L: m = L&15 ; k = 8*(L>>4) + 16*(elem>>3) + (elem&7)
// B-fragment uses the mirrored layout (lane = column n) -> works with weights
// stored transposed as [n][k].  Each lane reads two contiguous 16B chunks.
// ---------------------------------------------------------------------------
__device__ __forceinline__ bf16x16 load_frag(const u16* p, int row, int kbase, int lane) {
  const u16* q = p + (size_t)(row + (lane & 15)) * DD + (kbase + ((lane >> 4) << 3));
  union { uint4 u[2]; bf16x16 v; } f;
  f.u[0] = *(const uint4*)(q);
  f.u[1] = *(const uint4*)(q + 16);
  return f.v;
}

// C/D layout: lane L holds column n=L&15 ; VGPR j holds row m = 8*(L>>4)+j

// ---------------------------------------------------------------------------
// Weight transpose + f32->bf16 convert:  dst[h][n][k] = src[h*hstride + (roff+k)*DD + n]
// ---------------------------------------------------------------------------
__global__ void wcvt_kernel(const float* __restrict__ src, u16* __restrict__ dst,
                            int roff, int hstride) {
  int idx = blockIdx.x * blockDim.x + threadIdx.x;
  if (idx >= HH * DD * DD) return;
  int h = idx >> 16;
  int n = (idx >> 8) & 255;
  int k = idx & 255;
  dst[idx] = f2bf(src[(size_t)h * hstride + (size_t)(roff + k) * DD + n]);
}

// ---------------------------------------------------------------------------
// q = emb_r[batch_q] @ qp_W + qp_b ; also emit emb_q output
// ---------------------------------------------------------------------------
__global__ void qproj_kernel(const int* __restrict__ batch_q,
                             const float* __restrict__ emb_r,
                             const float* __restrict__ qpW,
                             const float* __restrict__ qpb,
                             float* __restrict__ qout,
                             float* __restrict__ emb_q_out) {
  __shared__ float rq[DD];
  int b = blockIdx.x, d = threadIdx.x;
  int qi = batch_q[b];
  rq[d] = emb_r[(size_t)qi * DD + d];
  __syncthreads();
  float s = qpb[d];
#pragma unroll 4
  for (int k = 0; k < DD; ++k) s += rq[k] * qpW[k * DD + d];
  qout[(size_t)b * DD + d] = s;
  emb_q_out[(size_t)b * DD + d] = rq[d];
}

// ---------------------------------------------------------------------------
// Fused attention layer: one workgroup per batch row b, 512 threads = 16 waves,
// wave w owns output-column strip [16w, 16w+16).  Flash-style softmax over
// 4 uniform chunks of 64 neighbor rows (rows >= 200 masked; over-read rows
// 208..255 land inside adjacent LDS regions that always hold finite bf16).
// LAYER=0 writes h_cat[B,1024]; LAYER=1 writes h+X into d_out.
// ---------------------------------------------------------------------------
// dynamic LDS layout (bytes)
#define OFF_SR   0                               // u16 [NT][DD]      106496
#define OFF_SE   (OFF_SR  + NT*DD*2)             // u16 [NT][DD]      106496
#define OFF_YB   (OFF_SE  + NT*DD*2)             // u16 [CH][DD]       32768
#define OFF_XB   (OFF_YB  + CH*DD*2)             // u16 [CH][DD]       32768
#define OFF_HT   (OFF_XB  + CH*DD*2)             // f32 [HH][DD]        4096
#define OFF_QS   (OFF_HT  + HH*DD*4)             // f32 [DD]            1024
#define OFF_HB   (OFF_QS  + DD*4)                // f32 [HH*DD]         4096
#define OFF_NEG  (OFF_HB  + HH*DD*4)             // f32 [256]           1024
#define OFF_LOG  (OFF_NEG + 256*4)               // f32 [CH]             256
#define OFF_PB   (OFF_LOG + CH*4)                // f32 [CH]             256
#define OFF_PART (OFF_PB  + CH*4)                // f32 [512]           2048
#define OFF_ACC  (OFF_PART+ 512*4)               // f32 [DD]            1024
#define OFF_HS   (OFF_ACC + DD*4)                // f32 [DD]            1024
#define OFF_SC   (OFF_HS  + DD*4)                // f32 [8]               32
#define SMEM_BYTES (OFF_SC + 32)

template<int LAYER>
__global__ __launch_bounds__(512, 1)
void attn_layer_kernel(const int*   __restrict__ batch_e1,
                       const int*   __restrict__ neighbors,   // [B,N,2] (rel, ent)
                       const float* __restrict__ masks,       // [B,N]
                       const float* __restrict__ emb_e,
                       const float* __restrict__ emb_r,
                       const float* __restrict__ fpW,         // [H, rows, D] f32
                       const float* __restrict__ fpB,         // [H, D]
                       const float* __restrict__ tB,          // [H, D]
                       const u16*   __restrict__ wrT,         // bf16 [H][D][D] (n,k)
                       const u16*   __restrict__ weT,
                       const u16*   __restrict__ twT,
                       const float* __restrict__ qbuf,        // [B, D]
                       const float* __restrict__ hcat_in,     // [B, H*D] (LAYER==1)
                       float*       __restrict__ outp) {      // L0: [B,H*D]  L1: [B,D]
  extern __shared__ char smem[];
  u16*   sR    = (u16*)  (smem + OFF_SR);
  u16*   sE    = (u16*)  (smem + OFF_SE);
  u16*   ybuf  = (u16*)  (smem + OFF_YB);
  u16*   xbuf  = (u16*)  (smem + OFF_XB);
  float* hterm = (float*)(smem + OFF_HT);
  float* qs    = (float*)(smem + OFF_QS);
  float* hbuf  = (float*)(smem + OFF_HB);
  float* negm  = (float*)(smem + OFF_NEG);
  float* logit = (float*)(smem + OFF_LOG);
  float* pbuf  = (float*)(smem + OFF_PB);
  float* part  = (float*)(smem + OFF_PART);
  float* accum = (float*)(smem + OFF_ACC);
  float* hsum  = (float*)(smem + OFF_HS);
  float* sc    = (float*)(smem + OFF_SC);

  const int b    = blockIdx.x;
  const int tid  = threadIdx.x;
  const int lane = tid & 31;
  const int wave = tid >> 5;               // 16 waves
  const int n0   = wave * 16;              // this wave's output-column strip

  constexpr int Kh   = (LAYER == 0) ? DD : HH * DD;          // h-part K
  constexpr int FPH  = (LAYER == 0) ? 3 * DD * DD : (HH + 2) * DD * DD;

  // ---- phase 0: q, h-input, head-mean init ------------------------------
  if (tid < DD) {
    qs[tid]   = qbuf[(size_t)b * DD + tid];
    hsum[tid] = 0.f;
    negm[tid] = (tid < NN) ? HUGE_I * (1.0f - masks[(size_t)b * NN + tid]) : HUGE_I;
  }
  for (int k = tid; k < Kh; k += 512) {
    if (LAYER == 0) hbuf[k] = emb_e[(size_t)batch_e1[b] * DD + k];
    else            hbuf[k] = hcat_in[(size_t)b * (HH * DD) + k];
  }
  __syncthreads();

  // ---- phase 1: hterm[h][d] = fp_b + hin @ Wh  (f32, small) -------------
  for (int oi = tid; oi < HH * DD; oi += 512) {
    int h = oi >> 8, d = oi & 255;
    float s = fpB[h * DD + d];
    const float* wp = fpW + (size_t)h * FPH + d;
#pragma unroll 4
    for (int k = 0; k < Kh; ++k) s += hbuf[k] * wp[(size_t)k * DD];
    hterm[oi] = s;
  }

  // ---- phase 2: gather neighbor embeddings -> LDS bf16 ------------------
  for (int idx = tid; idx < NT * DD; idx += 512) {
    int row = idx >> 8, col = idx & 255;
    u16 vr = 0, ve = 0;
    if (row < NN) {
      int rel = neighbors[((size_t)b * NN + row) * 2 + 0];
      int ent = neighbors[((size_t)b * NN + row) * 2 + 1];
      vr = f2bf(emb_r[(size_t)rel * DD + col]);
      ve = f2bf(emb_e[(size_t)ent * DD + col]);
    }
    sR[idx] = vr;
    sE[idx] = ve;
  }
  __syncthreads();

  // ---- phase 3: heads ---------------------------------------------------
  for (int h = 0; h < HH; ++h) {
    if (tid == 0) { sc[0] = -INFINITY; sc[1] = 0.f; }
    if (tid < DD) accum[tid] = 0.f;
    __syncthreads();

    const u16* wrh = wrT + (size_t)h * DD * DD;
    const u16* weh = weT + (size_t)h * DD * DD;
    const u16* twh = twT + (size_t)h * DD * DD;
    const float hinit = hterm[h * DD + n0 + (lane & 15)];
    const float tbv   = tB[h * DD + n0 + (lane & 15)];

    for (int c = 0; c < NCHUNK; ++c) {
      const int rows0 = c * CH;

      // ---- GEMM1: y = hterm + r@Wr + e@We (bf16 WMMA, f32 acc) ----------
      f32x8 acc[4];
#pragma unroll
      for (int t = 0; t < 4; ++t)
#pragma unroll
        for (int j = 0; j < 8; ++j) acc[t][j] = 0.f;

      for (int kk = 0; kk < DD; kk += 32) {
        bf16x16 brf = load_frag(wrh, n0, kk, lane);
        bf16x16 bef = load_frag(weh, n0, kk, lane);
#pragma unroll
        for (int t = 0; t < 4; ++t) {      // 4 independent WMMAs in a row
          bf16x16 ar = load_frag(sR, rows0 + t * 16, kk, lane);
          acc[t] = __builtin_amdgcn_wmma_f32_16x16x32_bf16(
              false, ar, false, brf, (short)0, acc[t], false, false);
        }
#pragma unroll
        for (int t = 0; t < 4; ++t) {      // dependent pair separated by 3 others
          bf16x16 ae = load_frag(sE, rows0 + t * 16, kk, lane);
          acc[t] = __builtin_amdgcn_wmma_f32_16x16x32_bf16(
              false, ae, false, bef, (short)0, acc[t], false, false);
        }
      }
#pragma unroll
      for (int t = 0; t < 4; ++t)
#pragma unroll
        for (int j = 0; j < 8; ++j) {
          int m = t * 16 + ((lane >> 4) << 3) + j;
          ybuf[m * DD + n0 + (lane & 15)] = f2bf(acc[t][j] + hinit);
        }
      __syncthreads();

      // ---- GEMM2: x = leaky(y @ tW + t_b, 0.1) --------------------------
      f32x8 acc2[4];
#pragma unroll
      for (int t = 0; t < 4; ++t)
#pragma unroll
        for (int j = 0; j < 8; ++j) acc2[t][j] = 0.f;

      for (int kk = 0; kk < DD; kk += 32) {
        bf16x16 bwf = load_frag(twh, n0, kk, lane);
#pragma unroll
        for (int t = 0; t < 4; ++t) {
          bf16x16 ay = load_frag(ybuf, t * 16, kk, lane);
          acc2[t] = __builtin_amdgcn_wmma_f32_16x16x32_bf16(
              false, ay, false, bwf, (short)0, acc2[t], false, false);
        }
      }
#pragma unroll
      for (int t = 0; t < 4; ++t)
#pragma unroll
        for (int j = 0; j < 8; ++j) {
          float v = acc2[t][j] + tbv;
          v = (v >= 0.f) ? v : 0.1f * v;
          int m = t * 16 + ((lane >> 4) << 3) + j;
          xbuf[m * DD + n0 + (lane & 15)] = f2bf(v);
        }
      __syncthreads();

      // ---- logits: x . q  (8 threads / row, all 64 rows) ----------------
      {
        int m = tid >> 3, j0 = (tid & 7) * 32;
        float p = 0.f;
#pragma unroll 8
        for (int i = 0; i < 32; ++i) p += bf2f(xbuf[m * DD + j0 + i]) * qs[j0 + i];
        part[tid] = p;
      }
      __syncthreads();
      if (tid < CH) {
        float s = 0.f;
#pragma unroll
        for (int i = 0; i < 8; ++i) s += part[tid * 8 + i];
        logit[tid] = s - negm[rows0 + tid];
      }
      __syncthreads();

      // ---- streaming softmax update -------------------------------------
      if (tid == 0) {
        float cm = -INFINITY;
        for (int m = 0; m < CH; ++m) cm = fmaxf(cm, logit[m]);
        float oldM = sc[0];
        float newM = fmaxf(oldM, cm);
        sc[2] = __expf(oldM - newM);     // rescale factor (0 on first chunk)
        sc[0] = newM;
      }
      __syncthreads();
      if (tid < CH) pbuf[tid] = __expf(logit[tid] - sc[0]);
      __syncthreads();
      if (tid == 0) {
        float ps = 0.f;
        for (int m = 0; m < CH; ++m) ps += pbuf[m];
        sc[1] = sc[1] * sc[2] + ps;
      }
      if (tid < DD) {
        float a = accum[tid] * sc[2];
        for (int m = 0; m < CH; ++m) a += pbuf[m] * bf2f(xbuf[m * DD + tid]);
        accum[tid] = a;
      }
      __syncthreads();
    } // chunks

    // ---- finalize head: agg = leaky(acc/S, 0.01) ------------------------
    if (tid < DD) {
      float a = accum[tid] / sc[1];
      a = (a >= 0.f) ? a : 0.01f * a;
      if (LAYER == 0) outp[(size_t)b * (HH * DD) + h * DD + tid] = a;
      else            hsum[tid] += a;
    }
    __syncthreads();
  } // heads

  if (LAYER == 1 && tid < DD) {
    float X = hsum[tid] * (1.0f / HH);
    X = (X >= 0.f) ? X : 0.01f * X;
    outp[(size_t)b * DD + tid] = emb_e[(size_t)batch_e1[b] * DD + tid] + X;
  }
}

// ---------------------------------------------------------------------------
// workspace layout (bytes)
// ---------------------------------------------------------------------------
#define WS_Q    0                                   // f32 [B,D]
#define WS_HCAT (WS_Q    + BB*DD*4)                 // f32 [B,H*D]
#define WS_WR0  (WS_HCAT + BB*HH*DD*4)              // bf16 [H][D][D] each
#define WS_WE0  (WS_WR0  + HH*DD*DD*2)
#define WS_TW0  (WS_WE0  + HH*DD*DD*2)
#define WS_WR1  (WS_TW0  + HH*DD*DD*2)
#define WS_WE1  (WS_WR1  + HH*DD*DD*2)
#define WS_TW1  (WS_WE1  + HH*DD*DD*2)

extern "C" void kernel_launch(void* const* d_in, const int* in_sizes, int n_in,
                              void* d_out, int out_size, void* d_ws, size_t ws_size,
                              hipStream_t stream) {
  const int*   batch_e1 = (const int*)  d_in[0];
  const int*   batch_q  = (const int*)  d_in[1];
  const int*   neighbors= (const int*)  d_in[2];
  const float* masks    = (const float*)d_in[3];
  const float* emb_e    = (const float*)d_in[4];
  const float* emb_r    = (const float*)d_in[5];
  const float* qp_W     = (const float*)d_in[6];
  const float* qp_b     = (const float*)d_in[7];
  const float* fp_W0    = (const float*)d_in[8];
  const float* fp_b0    = (const float*)d_in[9];
  const float* t_W0     = (const float*)d_in[10];
  const float* t_b0     = (const float*)d_in[11];
  const float* fp_W1    = (const float*)d_in[12];
  const float* fp_b1    = (const float*)d_in[13];
  const float* t_W1     = (const float*)d_in[14];
  const float* t_b1     = (const float*)d_in[15];

  float* out  = (float*)d_out;
  char*  ws   = (char*)d_ws;
  float* q    = (float*)(ws + WS_Q);
  float* hcat = (float*)(ws + WS_HCAT);
  u16* wr0 = (u16*)(ws + WS_WR0);
  u16* we0 = (u16*)(ws + WS_WE0);
  u16* tw0 = (u16*)(ws + WS_TW0);
  u16* wr1 = (u16*)(ws + WS_WR1);
  u16* we1 = (u16*)(ws + WS_WE1);
  u16* tw1 = (u16*)(ws + WS_TW1);

  const int wthreads = HH * DD * DD;             // 262144
  const int wblocks  = (wthreads + 255) / 256;
  // fp_W0 rows: [0,D)=h, [D,2D)=r, [2D,3D)=e ; fp_W1 rows: [0,HD)=h_, [HD,HD+D)=r, ...
  wcvt_kernel<<<wblocks, 256, 0, stream>>>(fp_W0, wr0, DD,        3 * DD * DD);
  wcvt_kernel<<<wblocks, 256, 0, stream>>>(fp_W0, we0, 2 * DD,    3 * DD * DD);
  wcvt_kernel<<<wblocks, 256, 0, stream>>>(t_W0,  tw0, 0,         DD * DD);
  wcvt_kernel<<<wblocks, 256, 0, stream>>>(fp_W1, wr1, HH * DD,       (HH + 2) * DD * DD);
  wcvt_kernel<<<wblocks, 256, 0, stream>>>(fp_W1, we1, (HH + 1) * DD, (HH + 2) * DD * DD);
  wcvt_kernel<<<wblocks, 256, 0, stream>>>(t_W1,  tw1, 0,         DD * DD);

  qproj_kernel<<<BB, DD, 0, stream>>>(batch_q, emb_r, qp_W, qp_b, q, out + (size_t)BB * DD);

  attn_layer_kernel<0><<<BB, 512, SMEM_BYTES, stream>>>(
      batch_e1, neighbors, masks, emb_e, emb_r,
      fp_W0, fp_b0, t_b0, wr0, we0, tw0, q, nullptr, hcat);

  attn_layer_kernel<1><<<BB, 512, SMEM_BYTES, stream>>>(
      batch_e1, neighbors, masks, emb_e, emb_r,
      fp_W1, fp_b1, t_b1, wr1, we1, tw1, q, hcat, out);
}
